// GPT_12283606468080
// MI455X (gfx1250) — matrix operand order repaired and checked by reading, not compiled, not run on hardware
//
#include <hip/hip_runtime.h>
#include <hip/hip_bf16.h>

// MI455X (gfx1250) GPT forward. Compute-bound (~0.7 TFLOP vs ~550MB of weights
// @ 23.3 TB/s), so every GEMM + attention goes through v_wmma_f32_16x16x32_bf16
// (wave32 WMMA). fp32 master buffers; fp32->bf16 conversion (v_cvt_pk_bf16_f32)
// while staging into LDS, where tiles are stored PRE-PERMUTED into the WMMA
// VGPR layout so each lane loads its fragment with two ds_load_b128 ops.
// Staging is two-phase (batched loads, then cvt+store) so global loads overlap
// instead of serializing on s_wait_loadcnt 0. Attention keeps its 16x64 O
// accumulator in registers (4x v8f, WMMA C layout); two-pass LDS softmax.

typedef __attribute__((ext_vector_type(16))) __bf16 v16bf;
typedef __attribute__((ext_vector_type(8)))  __bf16 v8bf;
typedef __attribute__((ext_vector_type(2)))  __bf16 v2bf;
typedef __attribute__((ext_vector_type(8)))  float  v8f;

union VFrag { v16bf v; v8bf h[2]; };

#define TT 2048
#define DD 1024
#define HH 16
#define KVH 4
#define HD 64
#define FF 4096
#define LL 8
#define VV 50304

__device__ __forceinline__ unsigned pack_bf2(float a, float b) {
  v2bf pk;                         // lowers to one v_cvt_pk_bf16_f32
  pk[0] = (__bf16)a;
  pk[1] = (__bf16)b;
  return __builtin_bit_cast(unsigned, pk);
}

__device__ __forceinline__ v16bf ld_frag(const __bf16* p) {
  VFrag u;
  const v8bf* vp = (const v8bf*)p;    // 16B-aligned rows -> 2x ds_load_b128
  u.h[0] = vp[0];
  u.h[1] = vp[1];
  return u.v;
}

// ---------------- embedding + rms ----------------
__global__ void k_embed_rms(const int* __restrict__ idx,
                            const float* __restrict__ wte,
                            float* __restrict__ x) {
  const int t = blockIdx.x;
  const int tid = threadIdx.x & 255;
  const float* src = wte + (long long)idx[t] * DD;
  __shared__ float red[256];
  const float4 v4 = *(const float4*)(src + tid * 4);
  red[tid] = v4.x * v4.x + v4.y * v4.y + v4.z * v4.z + v4.w * v4.w;
  __syncthreads();
  for (int s = 128; s > 0; s >>= 1) {
    if (tid < s) red[tid] += red[tid + s];
    __syncthreads();
  }
  const float r = rsqrtf(red[0] * (1.0f / DD) + 1e-6f);
  float4 o; o.x = v4.x * r; o.y = v4.y * r; o.z = v4.z * r; o.w = v4.w * r;
  *(float4*)(x + (long long)t * DD + tid * 4) = o;
}

// ---------------- rmsnorm over rows of width D ----------------
__global__ void k_rms(const float* __restrict__ in, float* __restrict__ out) {
  const int t = blockIdx.x;
  const int tid = threadIdx.x & 255;
  const float* src = in + (long long)t * DD;
  __shared__ float red[256];
  const float4 v4 = *(const float4*)(src + tid * 4);
  red[tid] = v4.x * v4.x + v4.y * v4.y + v4.z * v4.z + v4.w * v4.w;
  __syncthreads();
  for (int s = 128; s > 0; s >>= 1) {
    if (tid < s) red[tid] += red[tid + s];
    __syncthreads();
  }
  const float r = rsqrtf(red[0] * (1.0f / DD) + 1e-6f);
  float4 o; o.x = v4.x * r; o.y = v4.y * r; o.z = v4.z * r; o.w = v4.w * r;
  *(float4*)(out + (long long)t * DD + tid * 4) = o;
}

// ---------------- WMMA GEMM: C = epilogue(A[M,K] @ B[K,N]) ----------------
// Block: 256 thr = 8 waves; block tile 32x128; wave tile 16x32 (2 WMMA/K-step).
// EPI: 0 = store, 1 = residual add (C += AB), 2 = relu^2, 3 = softcap tanh
template <int EPI>
__global__ void k_gemm(const float* __restrict__ A, const float* __restrict__ B,
                       float* __restrict__ C, int M, int N, int K) {
  // Tiles stored in WMMA fragment layout: [tileIdx][lane][slot]
  __shared__ __align__(16) __bf16 Aperm[2][32][16];   // 32 rows x 32 k
  __shared__ __align__(16) __bf16 Bperm[8][32][16];   // 32 k x 128 cols
  const int tid  = threadIdx.x & 255;   // bounded ranges -> compiler can fold
  const int lane = tid & 31;
  const int wave = tid >> 5;
  const int hl   = lane >> 4;
  const int r    = lane & 15;
  const int wm   = wave >> 2;           // 0..1 : wave row (16 rows)
  const int wn   = wave & 3;            // 0..3 : wave col (32 cols = 2 tiles)
  const int rowBase = blockIdx.y * 32;
  const int colBase = blockIdx.x * 128;
  const float* __restrict__ Abase = A + (long long)rowBase * K;  // uniform
  const float* __restrict__ Bbase = B + colBase;                 // uniform

  // per-lane loop-invariant staging coordinates
  const int rA = tid >> 4;              // 0..15 : A row within 16 (+16*it)
  const int kA = (tid & 15) * 2;        // even k for A pair
  const int laneA = (kA >> 4) * 16 + rA;
  const int iiA   = kA & 15;
  const int cB = tid & 127;             // 0..127 : B column
  const int kB = ((tid >> 7) & 1) * 2;  // 0 or 2 : B k base (+4*it)

  v8f acc[2] = {{}, {}};
  for (int k0 = 0; k0 < K; k0 += 32) {
    // phase 1: batch all global loads for this K tile (stay in flight)
    float2 av[2];
    float  b0a[8], b1a[8];
#pragma unroll
    for (int it = 0; it < 2; ++it)
      av[it] = *(const float2*)(Abase + (rA + 16 * it) * K + k0 + kA);
#pragma unroll
    for (int it = 0; it < 8; ++it) {
      const int kk = kB + it * 4;
      b0a[it] = Bbase[(k0 + kk) * N + cB];
      b1a[it] = Bbase[(k0 + kk + 1) * N + cB];
    }
    if (k0 + 32 < K) {   // global_prefetch_b8 next K tile
      __builtin_prefetch(Abase + (rA + 16) * K + k0 + 32 + kA, 0, 1);
      __builtin_prefetch(Bbase + (k0 + 32 + kB) * N + cB, 0, 1);
    }
    // phase 2: convert + permuted LDS stores
#pragma unroll
    for (int it = 0; it < 2; ++it)
      ((unsigned*)Aperm)[(it * 512 + laneA * 16 + iiA) >> 1] =
          pack_bf2(av[it].x, av[it].y);
#pragma unroll
    for (int it = 0; it < 8; ++it) {
      const int kk = kB + it * 4;
      const int lane_ = (kk >> 4) * 16 + (cB & 15);
      ((unsigned*)Bperm)[((cB >> 4) * 512 + lane_ * 16 + (kk & 15)) >> 1] =
          pack_bf2(b0a[it], b1a[it]);
    }
    __syncthreads();

    const v16bf af = ld_frag(Aperm[wm][lane]);
#pragma unroll
    for (int nb = 0; nb < 2; ++nb) {
      const v16bf bv = ld_frag(Bperm[wn * 2 + nb][lane]);
      acc[nb] = __builtin_amdgcn_wmma_f32_16x16x32_bf16(
          false, af, false, bv, (short)0, acc[nb], false, false);
    }
    __syncthreads();
  }

  float* __restrict__ Cbase = C + (long long)rowBase * N + colBase;
#pragma unroll
  for (int nb = 0; nb < 2; ++nb) {
#pragma unroll
    for (int j = 0; j < 8; ++j) {          // C layout: M = j + 8*half, N = r
      const int o = (wm * 16 + j + hl * 8) * N + wn * 32 + nb * 16 + r;
      float val = acc[nb][j];
      if (EPI == 1) val += Cbase[o];
      if (EPI == 2) { val = fmaxf(val, 0.f); val = val * val; }
      if (EPI == 3) { val = 15.0f * tanhf(val * (1.0f / 15.0f)); }
      Cbase[o] = val;
    }
  }
}

// ---------------- RoPE + per-head RMS (one wave per (t, head)) ----------------
__global__ void k_rope_rms(float* __restrict__ buf, int heads) {
  const int t = blockIdx.x;
  const int h = blockIdx.y;
  const int lane = threadIdx.x & 31;   // 0..31 = frequency index
  float* p = buf + (long long)t * heads * HD + h * HD;
  const float x1 = p[lane];
  const float x2 = p[lane + 32];
  const float inv = __powf(10000.0f, -(float)lane * (1.0f / 32.0f));
  const float fr = (float)t * inv;
  float sn, cs;
  __sincosf(fr, &sn, &cs);
  const float o1 =  x1 * cs + x2 * sn;
  const float o2 = -x1 * sn + x2 * cs;
  float ss = o1 * o1 + o2 * o2;
#pragma unroll
  for (int m = 16; m > 0; m >>= 1) ss += __shfl_xor(ss, m, 32);
  const float rs = rsqrtf(ss * (1.0f / HD) + 1e-6f);
  p[lane]      = o1 * rs;
  p[lane + 32] = o2 * rs;
}

// ---------------- flash attention: one wave per (head, 16-query tile) --------
// O accumulator (16x64) lives in registers: 4x v8f in WMMA C layout.
__global__ void k_attn(const float* __restrict__ q, const float* __restrict__ k,
                       const float* __restrict__ v, float* __restrict__ y) {
  const int head = blockIdx.x;      // 0..15
  const int qb   = blockIdx.y;      // 0..127 (16 queries each)
  const int kvh  = head >> 2;       // GQA: 4 q heads per kv head
  const int lane = threadIdx.x & 31;
  const int hl = lane >> 4, r = lane & 15;

  __shared__ __align__(16) __bf16 Qperm[2][32][16];     // A-layout, kd=0/1
  __shared__ __align__(16) __bf16 Kperm[2][2][32][16];  // B-layout, [nb][kd]
  __shared__ __align__(16) __bf16 Vperm[4][32][16];     // B-layout, [nb2]
  __shared__ __align__(16) __bf16 Pperm[32][16];        // A-layout
  __shared__ float  Sb[16][32];
  __shared__ float  alpha[16];
  __shared__ float  linv[16];

  const float* __restrict__ qbase = q + qb * 16 * DD + head * HD;  // uniform
  const float* __restrict__ kbase = k + kvh * HD;                  // uniform
  const float* __restrict__ vbase = v + kvh * HD;                  // uniform

  // loop-invariant permute coords (d = 2*lane for Q/K staging)
  const int dl   = 2 * r;                      // d & 31
  const int laneQ = (dl >> 4) * 16;            // + it
  const int iiQ   = dl & 15;
  const int laneK = (dl >> 4) * 16;            // + (it & 15)
  const int iiK   = dl & 15;

  // stage Q (16x64): row = it, d = 2*lane; batched float2 loads
#pragma unroll
  for (int g = 0; g < 2; ++g) {
    float2 qv[8];
#pragma unroll
    for (int u = 0; u < 8; ++u)
      qv[u] = *(const float2*)(qbase + (g * 8 + u) * DD + 2 * lane);
#pragma unroll
    for (int u = 0; u < 8; ++u) {
      const int it = g * 8 + u;
      ((unsigned*)Qperm)[(hl * 512 + (laneQ + it) * 16 + iiQ) >> 1] =
          pack_bf2(qv[u].x, qv[u].y);
    }
  }
  v8f oacc[4] = {{}, {}, {}, {}};      // O accumulator, WMMA C layout
  float m_st = -3.0e38f, l_st = 0.f;   // meaningful in lanes 0..15
  __syncthreads();

  const int nkb = (qb >> 1) + 1;       // 32-key blocks covering causal range
  for (int kb = 0; kb < nkb; ++kb) {
    const int ks = kb * 32;
    // stage K: key = it, d = 2*lane; batched float2 loads
#pragma unroll
    for (int g = 0; g < 4; ++g) {
      float2 kv[8];
#pragma unroll
      for (int u = 0; u < 8; ++u)
        kv[u] = *(const float2*)(kbase + (ks + g * 8 + u) * (KVH * HD) + 2 * lane);
#pragma unroll
      for (int u = 0; u < 8; ++u) {
        const int it = g * 8 + u;
        ((unsigned*)Kperm)[(((it >> 4) * 2 + hl) * 512 + (laneK + (it & 15)) * 16 + iiK) >> 1] =
            pack_bf2(kv[u].x, kv[u].y);
      }
    }
    // stage V: kpair = it>>1, col = (it&1)*32 + lane; batched loads
#pragma unroll
    for (int g = 0; g < 4; ++g) {
      float v0a[8], v1a[8];
#pragma unroll
      for (int u = 0; u < 8; ++u) {
        const int it = g * 8 + u;
        const int kk = (it >> 1) * 2;
        const int c  = (it & 1) * 32 + lane;
        v0a[u] = vbase[(ks + kk) * (KVH * HD) + c];
        v1a[u] = vbase[(ks + kk + 1) * (KVH * HD) + c];
      }
#pragma unroll
      for (int u = 0; u < 8; ++u) {
        const int it = g * 8 + u;
        const int kk = (it >> 1) * 2;
        const int c  = (it & 1) * 32 + lane;
        ((unsigned*)Vperm)[((c >> 4) * 512 + ((kk >> 4) * 16 + (c & 15)) * 16 + (kk & 15)) >> 1] =
            pack_bf2(v0a[u], v1a[u]);
      }
    }
    __syncthreads();

    // S = (Q @ K^T) * scale : two 16x16 tiles, accumulated over HD=64
#pragma unroll
    for (int nb = 0; nb < 2; ++nb) {
      v8f sa = {};
#pragma unroll
      for (int kd = 0; kd < 2; ++kd) {
        const v16bf af = ld_frag(Qperm[kd][lane]);
        const v16bf bv = ld_frag(Kperm[nb][kd][lane]);
        sa = __builtin_amdgcn_wmma_f32_16x16x32_bf16(
            false, af, false, bv, (short)0, sa, false, false);
      }
#pragma unroll
      for (int j = 0; j < 8; ++j)
        Sb[j + hl * 8][nb * 16 + r] = sa[j] * 0.125f;   // 1/sqrt(64)
    }
    __syncthreads();

    // online softmax (two-pass over Sb, no big live arrays): lane = query row
    if (lane < 16) {
      const int nval = qb * 16 + lane - ks;   // valid keys: c2 <= nval
      float rm = -3.0e38f;
#pragma unroll
      for (int c2 = 0; c2 < 32; ++c2) {       // pass 1: row max (masked)
        const float s = Sb[lane][c2];
        rm = fmaxf(rm, (c2 <= nval) ? s : -3.0e38f);
      }
      const float nm = fmaxf(m_st, rm);
      const float al = __expf(m_st - nm);
      float ps = 0.f;
#pragma unroll
      for (int c2 = 0; c2 < 32; c2 += 2) {    // pass 2: exp + pack (masked)
        float p0 = __expf(Sb[lane][c2]     - nm);
        float p1 = __expf(Sb[lane][c2 + 1] - nm);
        if (c2     > nval) p0 = 0.f;
        if (c2 + 1 > nval) p1 = 0.f;
        ps += p0 + p1;
        const int lane_ = (c2 >> 4) * 16 + lane;
        const int ii = c2 & 15;
        ((unsigned*)Pperm)[(lane_ * 16 + ii) >> 1] = pack_bf2(p0, p1);
      }
      l_st = l_st * al + ps;
      m_st = nm;
      alpha[lane] = al;
    }
    __syncthreads();

    // per-row alpha for this lane's 8 C-layout rows (broadcast LDS reads)
    float ar[8];
#pragma unroll
    for (int j = 0; j < 8; ++j) ar[j] = alpha[j + hl * 8];

    // O = O*alpha + P @ V : four 16x16 column tiles, K=32, all in registers
    const v16bf afP = ld_frag(Pperm[lane]);
#pragma unroll
    for (int nb2 = 0; nb2 < 4; ++nb2) {
#pragma unroll
      for (int j = 0; j < 8; ++j) oacc[nb2][j] *= ar[j];
      const v16bf bv = ld_frag(Vperm[nb2][lane]);
      oacc[nb2] = __builtin_amdgcn_wmma_f32_16x16x32_bf16(
          false, afP, false, bv, (short)0, oacc[nb2], false, false);
    }
    __syncthreads();
  }

  if (lane < 16) linv[lane] = 1.0f / l_st;
  __syncthreads();
  float lv[8];
#pragma unroll
  for (int j = 0; j < 8; ++j) lv[j] = linv[j + hl * 8];
  float* __restrict__ ybase = y + qb * 16 * DD + head * HD;
#pragma unroll
  for (int nb2 = 0; nb2 < 4; ++nb2) {
#pragma unroll
    for (int j = 0; j < 8; ++j) {
      ybase[(j + hl * 8) * DD + nb2 * 16 + r] = oacc[nb2][j] * lv[j];
    }
  }
}

// ---------------- launcher ----------------
extern "C" void kernel_launch(void* const* d_in, const int* in_sizes, int n_in,
                              void* d_out, int out_size, void* d_ws, size_t ws_size,
                              hipStream_t stream) {
  (void)in_sizes; (void)n_in; (void)out_size; (void)ws_size;
  const int*   idx     = (const int*)d_in[0];
  const float* wte     = (const float*)d_in[1];
  const float* Wq      = (const float*)d_in[2];
  const float* Wk      = (const float*)d_in[3];
  const float* Wv      = (const float*)d_in[4];
  const float* Wo      = (const float*)d_in[5];
  const float* Wfc     = (const float*)d_in[6];
  const float* Wproj   = (const float*)d_in[7];
  const float* lm_head = (const float*)d_in[8];

  float* ws = (float*)d_ws;
  float* x  = ws;                                   // T*D
  float* h  = ws + (size_t)2 * 1024 * 1024;         // T*D
  float* qb = ws + (size_t)4 * 1024 * 1024;         // T*1024
  float* kb = ws + (size_t)6 * 1024 * 1024;         // T*256
  float* vb = ws + (size_t)6 * 1024 * 1024 + 512 * 1024;
  float* yb = ws + (size_t)7 * 1024 * 1024;         // T*D
  float* ub = ws + (size_t)9 * 1024 * 1024;         // T*F

  k_embed_rms<<<TT, 256, 0, stream>>>(idx, wte, x);

  for (int l = 0; l < LL; ++l) {
    k_rms<<<TT, 256, 0, stream>>>(x, h);
    k_gemm<0><<<dim3(DD / 128, TT / 32), 256, 0, stream>>>(
        h, Wq + (long long)l * DD * DD, qb, TT, DD, DD);
    k_gemm<0><<<dim3((KVH * HD) / 128, TT / 32), 256, 0, stream>>>(
        h, Wk + (long long)l * DD * (KVH * HD), kb, TT, KVH * HD, DD);
    k_gemm<0><<<dim3((KVH * HD) / 128, TT / 32), 256, 0, stream>>>(
        h, Wv + (long long)l * DD * (KVH * HD), vb, TT, KVH * HD, DD);
    k_rope_rms<<<dim3(TT, HH), 32, 0, stream>>>(qb, HH);
    k_rope_rms<<<dim3(TT, KVH), 32, 0, stream>>>(kb, KVH);
    k_attn<<<dim3(HH, TT / 16), 32, 0, stream>>>(qb, kb, vb, yb);
    k_gemm<1><<<dim3(DD / 128, TT / 32), 256, 0, stream>>>(
        yb, Wo + (long long)l * DD * DD, x, TT, DD, DD);
    k_rms<<<TT, 256, 0, stream>>>(x, h);
    k_gemm<2><<<dim3(FF / 128, TT / 32), 256, 0, stream>>>(
        h, Wfc + (long long)l * DD * FF, ub, TT, FF, DD);
    k_gemm<1><<<dim3(DD / 128, TT / 32), 256, 0, stream>>>(
        ub, Wproj + (long long)l * FF * DD, x, TT, DD, FF);
  }

  k_rms<<<TT, 256, 0, stream>>>(x, h);
  k_gemm<3><<<dim3(VV / 128, TT / 32), 256, 0, stream>>>(
      h, lm_head, (float*)d_out, TT, VV, DD);
}
